// KGAN_14903536517227
// MI455X (gfx1250) — matrix operand order repaired and checked by reading, not compiled, not run on hardware
//
#include <hip/hip_runtime.h>

#define DIM      64
#define NMEM     32
#define NREL     16
#define NBATCH   256
#define NHOP     2
#define WPB      4          // waves per block

typedef __attribute__((ext_vector_type(16))) _Float16 v16h;
typedef __attribute__((ext_vector_type(8)))  float    v8f;

// Build an A-operand tile of Rh = h .* r for one 16x32 (MxK) slab.
// Native 16-bit A layout: lane<16 holds row (lane), elems 0..7 = K dbase+0..7,
// elems 8..15 = K dbase+16..23; lane>=16 holds the same row's other K-halves
// (handled by caller passing dbase = 8*laneHalf).
static __device__ inline v16h make_rh_tile(const float* __restrict__ hrow,
                                           const float* __restrict__ rrow,
                                           int dbase) {
  v16h a;
#pragma unroll
  for (int s = 0; s < 2; ++s) {
    const int dd = dbase + 16 * s;
    const float4 h0 = *(const float4*)(hrow + dd);
    const float4 h1 = *(const float4*)(hrow + dd + 4);
    const float4 r0 = *(const float4*)(rrow + dd);
    const float4 r1 = *(const float4*)(rrow + dd + 4);
    a[8 * s + 0] = (_Float16)(h0.x * r0.x);
    a[8 * s + 1] = (_Float16)(h0.y * r0.y);
    a[8 * s + 2] = (_Float16)(h0.z * r0.z);
    a[8 * s + 3] = (_Float16)(h0.w * r0.w);
    a[8 * s + 4] = (_Float16)(h1.x * r1.x);
    a[8 * s + 5] = (_Float16)(h1.y * r1.y);
    a[8 * s + 6] = (_Float16)(h1.z * r1.z);
    a[8 * s + 7] = (_Float16)(h1.w * r1.w);
  }
  return a;
}

__global__ __launch_bounds__(WPB * 32) void kgan_wmma_kernel(
    const float* __restrict__ entity_emb,
    const float* __restrict__ relation_emb,
    const int*   __restrict__ items,
    const int*   __restrict__ mem_h,
    const int*   __restrict__ mem_r,
    const int*   __restrict__ mem_t,
    float*       __restrict__ out) {
  const int lane  = threadIdx.x & 31;
  const int wv    = threadIdx.x >> 5;
  const int tuple = blockIdx.x * WPB + wv;          // ((hop*B + b)*R + r)
  const int b     = (tuple / NREL) % NBATCH;
  const int lh    = lane >> 4;                      // lane half (0/1)
  const int lm    = lane & 15;

  const long mbase = (long)tuple * NMEM;            // memories_[hrt] flat base

  // ---- B operands from v = entity_emb[items[b]] (wave-uniform row) ----
  // B layout (32x16, 16-bit): lane holds column N=lm, elems j<8 = K 8*lh+j, j>=8 = K 16+8*lh+(j-8).
  const float4* vp = (const float4*)(entity_emb + (long)items[b] * DIM);
  v16h bv0, bv1;                                    // v[0..31], v[32..63]
#pragma unroll
  for (int q = 0; q < 4; ++q) {
    const float4 f = vp[4 * (q >> 1) + 2 * lh + (q & 1)];
    const float4 g = vp[8 + 4 * (q >> 1) + 2 * lh + (q & 1)];
    bv0[4 * q + 0] = (_Float16)f.x;  bv0[4 * q + 1] = (_Float16)f.y;
    bv0[4 * q + 2] = (_Float16)f.z;  bv0[4 * q + 3] = (_Float16)f.w;
    bv1[4 * q + 0] = (_Float16)g.x;  bv1[4 * q + 1] = (_Float16)g.y;
    bv1[4 * q + 2] = (_Float16)g.z;  bv1[4 * q + 3] = (_Float16)g.w;
  }

  // ---- tail-row bases; prefetch them while we do the score phase ----
  int trow[16];
#pragma unroll
  for (int j = 0; j < 16; ++j) {
    trow[j] = mem_t[mbase + 8 * lh + (j & 7) + 16 * (j >> 3)] * DIM;
    __builtin_prefetch(entity_emb + trow[j] + lm, 0, 3);   // global_prefetch_b8
  }

  // ---- scores: S(32) = Rh(32x64) * v, via 4 WMMAs (two m-halves x two K-chunks)
  const float* h0 = entity_emb   + (long)mem_h[mbase + lm]      * DIM;
  const float* r0 = relation_emb + (long)mem_r[mbase + lm]      * DIM;
  const float* h1 = entity_emb   + (long)mem_h[mbase + 16 + lm] * DIM;
  const float* r1 = relation_emb + (long)mem_r[mbase + 16 + lm] * DIM;

  const int d0 = 8 * lh;
  v8f s1 = {};   // rows m=0..15  (this lane sees m = 8*lh + i in s1[i])
  v8f s2 = {};   // rows m=16..31 (this lane sees m = 16 + 8*lh + i in s2[i])
  s1 = __builtin_amdgcn_wmma_f32_16x16x32_f16(false, make_rh_tile(h0, r0, d0),
                                              false, bv0, (short)0, s1, false, false);
  s1 = __builtin_amdgcn_wmma_f32_16x16x32_f16(false, make_rh_tile(h0, r0, d0 + 32),
                                              false, bv1, (short)0, s1, false, false);
  s2 = __builtin_amdgcn_wmma_f32_16x16x32_f16(false, make_rh_tile(h1, r1, d0),
                                              false, bv0, (short)0, s2, false, false);
  s2 = __builtin_amdgcn_wmma_f32_16x16x32_f16(false, make_rh_tile(h1, r1, d0 + 32),
                                              false, bv1, (short)0, s2, false, false);

  // ---- softmax over 32 slots (lane half holds 16; merge halves via shfl_xor 16)
  float mx = s1[0];
#pragma unroll
  for (int i = 0; i < 8; ++i) { mx = fmaxf(mx, s1[i]); mx = fmaxf(mx, s2[i]); }
  mx = fmaxf(mx, __shfl_xor(mx, 16, 32));

  float e1[8], e2[8], sum = 0.0f;
#pragma unroll
  for (int i = 0; i < 8; ++i) {
    e1[i] = __expf(s1[i] - mx);
    e2[i] = __expf(s2[i] - mx);
    sum += e1[i] + e2[i];
  }
  sum += __shfl_xor(sum, 16, 32);
  const float inv = 1.0f / sum;

  // probs land exactly in the A-operand layout for the tail matmul:
  // lane<16 needs K = {0..7, 16..23} = (s1 rows, s2 rows) of this half; lane>=16
  // needs K = {8..15, 24..31} — which is precisely what s1/s2 hold here.
  v16h pa;
#pragma unroll
  for (int i = 0; i < 8; ++i) {
    pa[i]     = (_Float16)(e1[i] * inv);
    pa[8 + i] = (_Float16)(e2[i] * inv);
  }

  // ---- out(64) = p(32) . T(32x64): 4 WMMAs, one per 16-wide d-block ----
  const long obase = (long)tuple * DIM;
#pragma unroll
  for (int db = 0; db < 4; ++db) {
    v16h tb;  // B operand: lane holds column d = db*16 + lm, elems j<8 = m 8*lh+j, j>=8 = m 16+8*lh+(j-8)
#pragma unroll
    for (int j = 0; j < 16; ++j)
      tb[j] = (_Float16)entity_emb[trow[j] + db * 16 + lm];
    v8f o = {};
    o = __builtin_amdgcn_wmma_f32_16x16x32_f16(false, pa, false, tb,
                                               (short)0, o, false, false);
    // All D rows are identical (A rows are a broadcast of p); lanes 0-15 hold
    // row M=0 in o[0] at column N=lane.
    if (lane < 16) out[obase + db * 16 + lm] = o[0];
  }
}

extern "C" void kernel_launch(void* const* d_in, const int* in_sizes, int n_in,
                              void* d_out, int out_size, void* d_ws, size_t ws_size,
                              hipStream_t stream) {
  (void)in_sizes; (void)n_in; (void)d_ws; (void)ws_size; (void)out_size;
  const float* entity_emb   = (const float*)d_in[0];
  const float* relation_emb = (const float*)d_in[1];
  const int*   items        = (const int*)d_in[2];
  const int*   mem_h        = (const int*)d_in[3];
  const int*   mem_r        = (const int*)d_in[4];
  const int*   mem_t        = (const int*)d_in[5];
  float* out = (float*)d_out;

  const int tuples = NHOP * NBATCH * NREL;          // 8192 waves total
  dim3 grid(tuples / WPB), block(WPB * 32);
  kgan_wmma_kernel<<<grid, block, 0, stream>>>(entity_emb, relation_emb, items,
                                               mem_h, mem_r, mem_t, out);
}
